// CausalSelfAttention_78743930405026
// MI455X (gfx1250) — matrix operand order repaired and checked by reading, not compiled, not run on hardware
//
#include <hip/hip_runtime.h>
#include <hip/hip_bf16.h>
#include <math.h>

// ---------------------------------------------------------------------------
// CausalSelfAttention (B=2, T=2048, C=2048, H=16, D=128) for gfx1250 (MI455X)
// Dense math on v_wmma_f32_16x16x32_bf16 (wave32 WMMA), fp32 accumulate.
// GEMM staging uses GLOBAL_LOAD_ASYNC_TO_LDS_B128 + ASYNCcnt double buffering.
// ---------------------------------------------------------------------------

#define B_   2
#define T_   2048
#define C_   2048
#define H_   16
#define D_   128

typedef __attribute__((ext_vector_type(16))) __bf16 v16bf;
typedef __attribute__((ext_vector_type(8)))  float  v8f;

union BfFrag { uint4 u[2]; v16bf v; };

// Two contiguous 16B loads -> one 16x32 (A) / 32x16 (B) bf16 WMMA fragment.
__device__ __forceinline__ v16bf load_frag(const __bf16* p0, const __bf16* p1) {
  BfFrag f;
  f.u[0] = *(const uint4*)p0;
  f.u[1] = *(const uint4*)p1;
  return f.v;
}

__device__ __forceinline__ v8f wmma_bf16(v16bf a, v16bf b, v8f c) {
  return __builtin_amdgcn_wmma_f32_16x16x32_bf16(
      /*neg_a=*/false, a, /*neg_b=*/false, b,
      /*c_mod=*/(short)0, c, /*reuse_a=*/false, /*reuse_b=*/false);
}

// Async global->LDS copy of 16 bytes per lane (CDNA5 async-tensor path).
__device__ __forceinline__ void async_copy_b128(void* lds_dst, const void* gsrc) {
  const unsigned lds32 = (unsigned)(size_t)lds_dst;  // low 32 bits = LDS offset
  const unsigned long long g64 = (unsigned long long)(size_t)gsrc;
  asm volatile("global_load_async_to_lds_b128 %0, %1, off"
               :: "v"(lds32), "v"(g64) : "memory");
}

// ---------------------------------------------------------------------------
// fp32 -> bf16 elementwise conversion
// ---------------------------------------------------------------------------
__global__ void cvt_f32_bf16_kernel(const float* __restrict__ in,
                                    __bf16* __restrict__ out, int n) {
  int i = blockIdx.x * blockDim.x + threadIdx.x;
  if (i < n) out[i] = (__bf16)in[i];
}

// ---------------------------------------------------------------------------
// C(MxN, OutT) = A(MxK, bf16) x B(KxN, bf16)
// 128x128 block, BK=32, 256 threads (8 waves), each wave owns a 32x64 patch
// (8 accumulators). LDS double-buffered, filled by async global->LDS loads.
// A-frag: lane holds row (lane&15), K chunks [8h,8h+8) and [16+8h,24+8h).
// B-frag: lane holds N column (lane&15), K rows [8h,8h+8) and [16+8h,24+8h).
// ---------------------------------------------------------------------------
template <typename OutT>
__global__ __launch_bounds__(256) void gemm_bf16_kernel(
    const __bf16* __restrict__ A, const __bf16* __restrict__ Bw,
    OutT* __restrict__ C, int M, int N, int K) {
  __shared__ alignas(16) __bf16 As[2][128 * 32];  // 8KB per buffer
  __shared__ alignas(16) __bf16 Bs[2][32 * 128];  // 8KB per buffer

  const int tid  = threadIdx.x;
  const int lane = tid & 31;
  const int w    = tid >> 5;    // wave 0..7
  const int wr   = w >> 1;      // 0..3: 32-row block
  const int wc   = w & 1;       // 0..1: 64-col block
  const int l15  = lane & 15;
  const int half = lane >> 4;

  const int m0 = blockIdx.y * 128;
  const int n0 = blockIdx.x * 128;

  v8f acc[2][4];
#pragma unroll
  for (int a = 0; a < 2; ++a)
#pragma unroll
    for (int nb = 0; nb < 4; ++nb) acc[a][nb] = (v8f){};

  // Stage one 128x32 A tile + 32x128 B tile: 512 16B-chunks each,
  // 2 A-chunks + 2 B-chunks per thread, issued as async LDS loads.
  auto stage = [&](int k0, int buf) {
#pragma unroll
    for (int i = 0; i < 2; ++i) {
      const int cid = tid + 256 * i;
      // A: chunk cid -> row cid>>2, K-chunk cid&3; LDS byte offset cid*16
      const int arow = cid >> 2, ach = cid & 3;
      async_copy_b128(&As[buf][cid * 8],
                      &A[(size_t)(m0 + arow) * K + k0 + ach * 8]);
      // B: chunk cid -> K-row cid>>4, N-chunk cid&15; LDS byte offset cid*16
      const int bkk = cid >> 4, bnch = cid & 15;
      async_copy_b128(&Bs[buf][cid * 8],
                      &Bw[(size_t)(k0 + bkk) * N + n0 + bnch * 8]);
    }
  };

  const int nk = K / 32;
  stage(0, 0);

  for (int kt = 0; kt < nk; ++kt) {
    const int buf = kt & 1;
    if (kt + 1 < nk) {
      stage((kt + 1) * 32, buf ^ 1);
      // 4 newest async ops pending; <=4 means current buffer's 4 are done.
      asm volatile("s_wait_asynccnt 0x4" ::: "memory");
    } else {
      asm volatile("s_wait_asynccnt 0x0" ::: "memory");
    }
    __syncthreads();  // all waves' current-buffer data visible

    v16bf af[2];
#pragma unroll
    for (int a = 0; a < 2; ++a) {
      const __bf16* ap = &As[buf][(32 * wr + 16 * a + l15) * 32];
      af[a] = load_frag(ap + 8 * half, ap + 16 + 8 * half);
    }
#pragma unroll
    for (int nb = 0; nb < 4; ++nb) {
      const __bf16* bp = &Bs[buf][64 * wc + 16 * nb + l15];
      v16bf bfr;
#pragma unroll
      for (int s = 0; s < 16; ++s)
        bfr[s] = bp[(8 * half + s + (s >= 8 ? 8 : 0)) * 128];
      acc[0][nb] = wmma_bf16(af[0], bfr, acc[0][nb]);
      acc[1][nb] = wmma_bf16(af[1], bfr, acc[1][nb]);
    }
    __syncthreads();  // reads of `buf` done before it is restaged next iter
  }

  // C layout: VGPR r -> row r + 8*half, col = lane&15 within each 16-col tile.
#pragma unroll
  for (int a = 0; a < 2; ++a) {
#pragma unroll
    for (int nb = 0; nb < 4; ++nb) {
      const int col = n0 + 64 * wc + 16 * nb + l15;
#pragma unroll
      for (int r = 0; r < 8; ++r) {
        const int row = m0 + 32 * wr + 16 * a + r + 8 * half;
        C[(size_t)row * N + col] = (OutT)acc[a][nb][r];
      }
    }
  }
}

// ---------------------------------------------------------------------------
// QK RMSNorm (fp32 math) + relayout:
//   q: (B,T,H,D) -> qh (B,H,T,D)   row-major  (A-fragment friendly)
//   k: (B,T,H,D) -> kT (B,H,D,T)   transposed (score B-fragment friendly)
//   v: (B,T,H,D) -> vh (B,H,T,D)   row-major  (P*V B-fragment friendly)
// One wave per (b,t,h) row of 128.
// ---------------------------------------------------------------------------
__global__ __launch_bounds__(256) void qkv_prep_kernel(
    const __bf16* __restrict__ qb, const __bf16* __restrict__ kb,
    const __bf16* __restrict__ vb, const float* __restrict__ qw,
    const float* __restrict__ kw, __bf16* __restrict__ qh,
    __bf16* __restrict__ kT, __bf16* __restrict__ vh) {
  const int lane = threadIdx.x & 31;
  const int row  = (blockIdx.x * 256 + threadIdx.x) >> 5;  // (b,t,h) flat
  const int h    = row & (H_ - 1);
  const int t    = (row >> 4) & (T_ - 1);
  const int b    = row >> 15;

  const size_t src = (size_t)row * D_;
  float qv[4], kv[4];
  float ssq = 0.f, ssk = 0.f;
#pragma unroll
  for (int j = 0; j < 4; ++j) {
    const int d = lane + 32 * j;
    qv[j] = (float)qb[src + d];
    kv[j] = (float)kb[src + d];
    ssq += qv[j] * qv[j];
    ssk += kv[j] * kv[j];
  }
#pragma unroll
  for (int mask = 16; mask >= 1; mask >>= 1) {
    ssq += __shfl_xor(ssq, mask);
    ssk += __shfl_xor(ssk, mask);
  }
  const float sq = rsqrtf(ssq * (1.0f / D_) + 1e-5f);
  const float sk = rsqrtf(ssk * (1.0f / D_) + 1e-5f);

  const size_t bh  = (size_t)(b * H_ + h);
  const size_t dst = (bh * T_ + t) * D_;
#pragma unroll
  for (int j = 0; j < 4; ++j) {
    const int d = lane + 32 * j;
    qh[dst + d] = (__bf16)(qv[j] * sq * qw[d]);
    kT[(bh * D_ + d) * T_ + t] = (__bf16)(kv[j] * sk * kw[d]);
    vh[dst + d] = vb[src + d];
  }
}

// ---------------------------------------------------------------------------
// Flash attention: one wave per (b, h, 16 query rows). 32-key tiles.
// Scores via 8 WMMAs (K=32 over D=128, two 16-key subtiles), ALiBi + causal
// applied in C-register layout, online softmax with intra-half shuffles,
// P re-shaped C->A layout through 1KB LDS, P*V via 8 WMMAs (d chunks of 16).
// Epilogue fuses 1/l and subln RMSNorm; writes bf16 y in (B,T,C).
// ---------------------------------------------------------------------------
__global__ __launch_bounds__(32) void attn_kernel(
    const __bf16* __restrict__ qh, const __bf16* __restrict__ kT,
    const __bf16* __restrict__ vh, const float* __restrict__ subw,
    __bf16* __restrict__ yb) {
  __shared__ alignas(16) __bf16 Plds[16 * 32];

  const int lane = threadIdx.x;
  const int half = lane >> 4;
  const int l15  = lane & 15;
  const int i0   = blockIdx.x * 16;
  const int h    = blockIdx.y;
  const int b    = blockIdx.z;
  const size_t bh = (size_t)(b * H_ + h);

  const float scale = 0.08838834764831845f;          // 1/sqrt(128)
  const float slope = exp2f(-0.5f * (float)(h + 1)); // ALiBi slope, H=16

  // Q fragments: 4 chunks of D (K=32 each)
  v16bf qf[4];
  {
    const __bf16* qp = qh + (bh * T_ + i0 + l15) * D_;
#pragma unroll
    for (int c = 0; c < 4; ++c)
      qf[c] = load_frag(qp + 32 * c + 8 * half, qp + 32 * c + 16 + 8 * half);
  }

  v8f O[8];
  float m_i[8], l_i[8];
#pragma unroll
  for (int c = 0; c < 8; ++c) O[c] = (v8f){};
#pragma unroll
  for (int r = 0; r < 8; ++r) { m_i[r] = -__builtin_inff(); l_i[r] = 0.f; }

  for (int j0 = 0; j0 <= i0 + 15; j0 += 32) {
    v8f S0 = {};
    v8f S1 = {};
#pragma unroll
    for (int c = 0; c < 4; ++c) {
      // B-frag: lane = key column (j0+l15); 16 d values in table order from kT
      const __bf16* kr = kT + (bh * D_ + 32 * c) * T_ + j0 + l15;
      v16bf bk0, bk1;
#pragma unroll
      for (int s = 0; s < 16; ++s) {
        const int kk = 8 * half + s + (s >= 8 ? 8 : 0);
        bk0[s] = kr[(size_t)kk * T_];
        bk1[s] = kr[(size_t)kk * T_ + 16];
      }
      S0 = wmma_bf16(qf[c], bk0, S0);
      S1 = wmma_bf16(qf[c], bk1, S1);
    }

    // bias + causal mask + online softmax (per-row stats, rows r+8*half)
#pragma unroll
    for (int r = 0; r < 8; ++r) {
      const int i  = i0 + r + 8 * half;
      const int ja = j0 + l15;
      const int jb = ja + 16;
      float s0 = (ja <= i) ? S0[r] * scale - slope * (float)(i - ja)
                           : -__builtin_inff();
      float s1 = (jb <= i) ? S1[r] * scale - slope * (float)(i - jb)
                           : -__builtin_inff();
      float rm = fmaxf(s0, s1);
#pragma unroll
      for (int mask = 8; mask >= 1; mask >>= 1)
        rm = fmaxf(rm, __shfl_xor(rm, mask));
      const float nm = fmaxf(m_i[r], rm);
      const float f  = __expf(m_i[r] - nm);
      m_i[r] = nm;
      const float p0 = __expf(s0 - nm);
      const float p1 = __expf(s1 - nm);
      float rs = p0 + p1;
#pragma unroll
      for (int mask = 8; mask >= 1; mask >>= 1) rs += __shfl_xor(rs, mask);
      l_i[r] = l_i[r] * f + rs;
#pragma unroll
      for (int c = 0; c < 8; ++c) O[c][r] *= f;
      const int m = r + 8 * half;
      Plds[m * 32 + l15]      = (__bf16)p0;
      Plds[m * 32 + 16 + l15] = (__bf16)p1;
    }
    __syncthreads();  // single-wave WG: DS in-order, barrier folds to NOP

    // P as A-fragment (16 rows x 32 keys)
    const __bf16* pp = &Plds[l15 * 32];
    v16bf pf = load_frag(pp + 8 * half, pp + 16 + 8 * half);

#pragma unroll
    for (int c = 0; c < 8; ++c) {
      // B-frag: lane = d column (16c+l15); 16 keys in table order from vh
      const __bf16* vr = vh + (bh * T_ + j0) * D_ + 16 * c + l15;
      v16bf vf;
#pragma unroll
      for (int s = 0; s < 16; ++s)
        vf[s] = vr[(size_t)(8 * half + s + (s >= 8 ? 8 : 0)) * D_];
      O[c] = wmma_bf16(pf, vf, O[c]);
    }
    __syncthreads();
  }

  // Epilogue: 1/l, subln RMSNorm over D, store y (B,T,C) bf16
#pragma unroll
  for (int r = 0; r < 8; ++r) {
    const float inv = 1.0f / l_i[r];
    float ss = 0.f;
#pragma unroll
    for (int c = 0; c < 8; ++c) {
      const float v = O[c][r] * inv;
      ss += v * v;
    }
#pragma unroll
    for (int mask = 8; mask >= 1; mask >>= 1) ss += __shfl_xor(ss, mask);
    const float rs = rsqrtf(ss * (1.0f / D_) + 1e-5f);
    const int t = i0 + r + 8 * half;
    const size_t dst = (((size_t)b * T_ + t) * H_ + h) * D_;
#pragma unroll
    for (int c = 0; c < 8; ++c) {
      const int d = 16 * c + l15;
      yb[dst + d] = (__bf16)(O[c][r] * inv * rs * subw[d]);
    }
  }
}

// ---------------------------------------------------------------------------
// Launcher
// ---------------------------------------------------------------------------
extern "C" void kernel_launch(void* const* d_in, const int* in_sizes, int n_in,
                              void* d_out, int out_size, void* d_ws,
                              size_t ws_size, hipStream_t stream) {
  (void)in_sizes; (void)n_in; (void)out_size; (void)ws_size;

  const float* x    = (const float*)d_in[0];
  const float* Wq   = (const float*)d_in[1];
  const float* Wk   = (const float*)d_in[2];
  const float* Wv   = (const float*)d_in[3];
  const float* Wp   = (const float*)d_in[4];
  const float* qrw  = (const float*)d_in[5];
  const float* krw  = (const float*)d_in[6];
  const float* subw = (const float*)d_in[7];
  float* out = (float*)d_out;

  uint8_t* ws = (uint8_t*)d_ws;
  const size_t NX = (size_t)B_ * T_ * C_;  // 8388608
  const size_t NW = (size_t)C_ * C_;       // 4194304
  __bf16* xb  = (__bf16*)(ws);
  __bf16* Wqb = (__bf16*)(ws + 2 * NX);
  __bf16* Wkb = (__bf16*)(ws + 2 * NX + 2 * NW);
  __bf16* Wvb = (__bf16*)(ws + 2 * NX + 4 * NW);
  __bf16* Wpb = (__bf16*)(ws + 2 * NX + 6 * NW);
  __bf16* qb  = (__bf16*)(ws + 2 * NX + 8 * NW);
  __bf16* kb  = (__bf16*)(ws + 4 * NX + 8 * NW);
  __bf16* vb  = (__bf16*)(ws + 6 * NX + 8 * NW);
  __bf16* qhp = (__bf16*)(ws + 8 * NX + 8 * NW);
  __bf16* kTp = (__bf16*)(ws + 10 * NX + 8 * NW);
  __bf16* vhp = (__bf16*)(ws + 12 * NX + 8 * NW);
  __bf16* yb  = qb;  // reuse: qb/kb/vb dead after qkv_prep

  const int M = B_ * T_;  // 4096

  cvt_f32_bf16_kernel<<<(int)((NX + 255) / 256), 256, 0, stream>>>(x, xb, (int)NX);
  cvt_f32_bf16_kernel<<<(int)((NW + 255) / 256), 256, 0, stream>>>(Wq, Wqb, (int)NW);
  cvt_f32_bf16_kernel<<<(int)((NW + 255) / 256), 256, 0, stream>>>(Wk, Wkb, (int)NW);
  cvt_f32_bf16_kernel<<<(int)((NW + 255) / 256), 256, 0, stream>>>(Wv, Wvb, (int)NW);
  cvt_f32_bf16_kernel<<<(int)((NW + 255) / 256), 256, 0, stream>>>(Wp, Wpb, (int)NW);

  dim3 gg(C_ / 128, M / 128);  // (16, 32)
  gemm_bf16_kernel<__bf16><<<gg, 256, 0, stream>>>(xb, Wqb, qb, M, C_, C_);
  gemm_bf16_kernel<__bf16><<<gg, 256, 0, stream>>>(xb, Wkb, kb, M, C_, C_);
  gemm_bf16_kernel<__bf16><<<gg, 256, 0, stream>>>(xb, Wvb, vb, M, C_, C_);

  qkv_prep_kernel<<<(B_ * T_ * H_) / 8, 256, 0, stream>>>(qb, kb, vb, qrw, krw,
                                                          qhp, kTp, vhp);

  attn_kernel<<<dim3(T_ / 16, H_, B_), 32, 0, stream>>>(qhp, kTp, vhp, subw, yb);

  gemm_bf16_kernel<float><<<gg, 256, 0, stream>>>(yb, Wpb, out, M, C_, C_);
}
